// ConvGraph_34273839022711
// MI455X (gfx1250) — matrix operand order repaired and checked by reading, not compiled, not run on hardware
//
#include <hip/hip_runtime.h>
#include <hip/hip_bf16.h>

// GCN layer for MI455X (gfx1250):
//   h = x @ W          (bf16 WMMA 16x16x32, W staged transposed in LDS,
//                       f32->bf16x2 packing via v_perm_b32)
//   out = A @ h        (COO scatter-add: 1 wave per edge, float4 per lane,
//                       L2-resident gather + global_atomic_add_f32 + prefetch)
// d_ws holds h: 100000*128*4 = 51.2 MB.

typedef __bf16 bf16x16 __attribute__((ext_vector_type(16)));
typedef float v8f     __attribute__((ext_vector_type(8)));

// Pack two f32 into a bf16x2 dword: add 0x8000 (round-half-up; matches RNE
// except at exact ties) then v_perm_b32 selects the two high halves.
// selector 0x07060302: b0=lo[2], b1=lo[3], b2=hi[2], b3=hi[3].
__device__ __forceinline__ unsigned pack_bf16(float lo, float hi) {
    unsigned ul = __builtin_bit_cast(unsigned, lo) + 0x8000u;
    unsigned uh = __builtin_bit_cast(unsigned, hi) + 0x8000u;
    return __builtin_amdgcn_perm(uh, ul, 0x07060302u);
}

// ---------------------------------------------------------------------------
// Kernel 1: zero the output (harness poisons d_out with 0xAA).
// ---------------------------------------------------------------------------
__global__ __launch_bounds__(256) void zero_f32(float* __restrict__ p, int n4) {
    int i = blockIdx.x * blockDim.x + threadIdx.x;
    float4 z = make_float4(0.f, 0.f, 0.f, 0.f);
    for (; i < n4; i += gridDim.x * blockDim.x)
        ((float4*)p)[i] = z;
}

// ---------------------------------------------------------------------------
// Kernel 2: h[M,128] = x[M,256] @ W[256,128] via v_wmma_f32_16x16x32_bf16.
// Block = 256 threads = 8 waves. Block tile = 64 rows x 128 cols.
// Wave w owns column tile 16*w; loops over 4 row tiles; K loop = 8 steps of 32.
// W is converted to bf16 and stored TRANSPOSED in LDS ([col][k], 64 KB) so the
// B fragment's (k, k+1) pairs are single 32-bit LDS reads.
// ---------------------------------------------------------------------------
__global__ __launch_bounds__(256) void gemm_bf16_wmma(
    const float* __restrict__ x, const float* __restrict__ W,
    float* __restrict__ h, int M) {
    __shared__ unsigned short wt[128 * 256];   // W^T in bf16, 64 KB
    unsigned* wtu = (unsigned*)wt;

    const int tid = threadIdx.x;
    // Stage W^T -> LDS as packed k-pairs. i enumerates (k-pair, column);
    // consecutive threads read consecutive columns of the same W row pair
    // (coalesced); one u32 LDS store per pair.
    for (int i = tid; i < 128 * 128; i += 256) {
        int n  = i & 127;          // column
        int kp = i >> 7;           // k-pair index, k = 2*kp
        float a = W[(2 * kp) * 128 + n];
        float b = W[(2 * kp + 1) * 128 + n];
        wtu[(n << 7) + kp] = pack_bf16(a, b);   // wt[n*256 + 2*kp .. +1]
    }
    __syncthreads();

    const int lane = tid & 31;
    const int wave = tid >> 5;
    const int half = lane >> 4;        // K-half selector for A/B fragments
    const int mn   = lane & 15;        // M index (A) / N index (B, C/D)
    const int col0 = wave * 16;
    const int rowBase = blockIdx.x * 64;

    for (int rt = 0; rt < 4; ++rt) {
        int row  = rowBase + rt * 16 + mn;
        int rowc = row < M ? row : (M - 1);             // clamp; store is guarded
        const float* xr = x + (size_t)rowc * 256;

        v8f c = {};
        for (int kb = 0; kb < 256; kb += 32) {
            union { bf16x16 v; unsigned u[8]; } a, b;
            // A fragment (16x32 bf16): vgpr v holds
            //   K = kb + 16*(v>=4) + 8*half + 2*(v&3) .. +1
            #pragma unroll
            for (int v = 0; v < 8; ++v) {
                int kk = kb + ((v & 4) << 2) + (half << 3) + ((v & 3) << 1);
                float2 f = *(const float2*)(xr + kk);   // 8B-aligned (kk even)
                a.u[v] = pack_bf16(f.x, f.y);
            }
            // B fragment (32x16 bf16): lane n = col; K order matches A; 2 K/VGPR
            #pragma unroll
            for (int v = 0; v < 8; ++v) {
                int kk = kb + ((v & 4) << 2) + (half << 3) + ((v & 3) << 1);
                b.u[v] = wtu[(((col0 + mn) << 8) + kk) >> 1];
            }
            // (neg_a, A, neg_b, B, c_mod, C, reuse_a, reuse_b)
            c = __builtin_amdgcn_wmma_f32_16x16x32_bf16(
                    false, a.v, false, b.v, (short)0, c, false, false);
        }

        // C/D layout: N = lane&15, M = vgpr + 8*half
        int outRowBase = rowBase + rt * 16 + (half << 3);
        float* hp = h + (size_t)outRowBase * 128 + col0 + mn;
        #pragma unroll
        for (int r = 0; r < 8; ++r) {
            if (outRowBase + r < M) hp[(size_t)r * 128] = c[r];
        }
    }
}

// ---------------------------------------------------------------------------
// Kernel 3: COO scatter-add. One wave per edge iteration: lane l covers
// columns 4l..4l+3 (32 lanes x float4 = one 128-float row). h and out both
// fit in the 192 MB L2, so gathers and atomics stay on-chip. Prefetch the
// next edge's source row one iteration ahead (global_prefetch_b8).
// ---------------------------------------------------------------------------
__global__ __launch_bounds__(256) void edge_scatter(
    const float* __restrict__ h,
    const int* __restrict__ edge_row, const int* __restrict__ edge_col,
    const float* __restrict__ edge_val, float* __restrict__ out, int E) {
    const int lane   = threadIdx.x & 31;
    const int wave   = blockIdx.x * (blockDim.x >> 5) + (threadIdx.x >> 5);
    const int nwaves = gridDim.x * (blockDim.x >> 5);

    for (int e = wave; e < E; e += nwaves) {
        int   dst = edge_row[e];
        int   src = edge_col[e];
        float val = edge_val[e];

        // Prefetch next iteration's gather row into cache.
        int en = e + nwaves;
        if (en < E) {
            int sn = edge_col[en];
            __builtin_prefetch(h + (size_t)sn * 128 + lane * 4, 0, 0);
        }

        float4 v = ((const float4*)(h + (size_t)src * 128))[lane];
        float* o = out + (size_t)dst * 128 + lane * 4;
        atomicAdd(o + 0, val * v.x);
        atomicAdd(o + 1, val * v.y);
        atomicAdd(o + 2, val * v.z);
        atomicAdd(o + 3, val * v.w);
    }
}

// ---------------------------------------------------------------------------
extern "C" void kernel_launch(void* const* d_in, const int* in_sizes, int n_in,
                              void* d_out, int out_size, void* d_ws, size_t ws_size,
                              hipStream_t stream) {
    const float* x        = (const float*)d_in[0];   // [M, 256]
    const float* W        = (const float*)d_in[1];   // [256, 128]
    const int*   edge_row = (const int*)  d_in[2];   // [E]
    const int*   edge_col = (const int*)  d_in[3];   // [E]
    const float* edge_val = (const float*)d_in[4];   // [E]
    float*       out      = (float*)d_out;           // [M, 128]
    float*       h        = (float*)d_ws;            // [M, 128] scratch (51.2 MB)

    const int M = in_sizes[0] / 256;
    const int E = in_sizes[2];

    // 1) out = 0
    {
        int n4 = (M * 128) / 4;
        int blocks = (n4 + 255) / 256;
        if (blocks > 4096) blocks = 4096;
        zero_f32<<<blocks, 256, 0, stream>>>(out, n4);
    }
    // 2) h = x @ W  (WMMA bf16)
    {
        int blocks = (M + 63) / 64;
        gemm_bf16_wmma<<<blocks, 256, 0, stream>>>(x, W, h, M);
    }
    // 3) out += A @ h (scatter-add)
    {
        edge_scatter<<<2048, 256, 0, stream>>>(h, edge_row, edge_col, edge_val,
                                               out, E);
    }
}
